// ObjectDetector_15642270892526
// MI455X (gfx1250) — compile-verified, matrix operand order checked by reading, not run on hardware
//
#include <hip/hip_runtime.h>
#include <hip/hip_bf16.h>

#define N_ROIS        10000
#define PAD_N         10048   // 157 * 64 : two 16x16x32 WMMA tiles per loop step, no tail guard
#define PAD_CNT       48.0f   // zero-bit pads land in bucket 0 of pass 7 only; corrected exactly
#define N_CLS         151
#define NMS_THRESH    0.3f
#define SCORE_THRESH  0.05f
#define MAX_PER_IMG   64
#define PRE_NMS_TOPN  6000
#define POST_NMS_TOPN 300

typedef __attribute__((ext_vector_type(16))) _Float16 v16h;
typedef __attribute__((ext_vector_type(8)))  float    v8f;

#define NEG_INF (-__builtin_inff())

// ---- dynamic-LDS layout (bytes); total ~236KB of the 320KB/WGP CDNA5 budget ----
#define SC_OFF    0        // 10048 f32 scores (working copy; -inf == inactive; zero pad)
#define BX_OFF    40192    // 10000 x 4 f32 boxes (16B aligned for ds_load_b128)
#define AR_OFF    200192   // 10000 f32 precomputed areas
#define KEEP_OFF  240192   // 313 u32 keep bitmask
#define HIST_OFF  241444   // 16 f32 radix histogram
#define REDV_OFF  241508   // 8 f32 per-wave reduction values
#define REDI_OFF  241540   // 8 i32 per-wave reduction indices
#define CTRL_OFF  241572   // u32[8]: 0=prefix 1=remaining(f32) 2=bestv(f32) 3=besti(i32) 4=colmax bits
#define LDS_BYTES 241664

__device__ __forceinline__ _Float16 ind_eq(float f, int shift, unsigned target) {
    return ((__float_as_uint(f) >> shift) == target) ? (_Float16)1.0f : (_Float16)0.0f;
}

// =====================================================================
// Kernel 1: one workgroup (256 threads = 8 wave32) per class.
//   stage -> WMMA radix-select top-6000 threshold -> fused greedy NMS -> write column
// =====================================================================
__global__ __launch_bounds__(256) void nms_per_class(const float* __restrict__ scores,
                                                     const float* __restrict__ boxes,
                                                     float* __restrict__ dists_out)
{
    extern __shared__ char smem[];
    float*    sc    = (float*)(smem + SC_OFF);
    float*    bxf   = (float*)(smem + BX_OFF);
    float4*   bx4   = (float4*)(smem + BX_OFF);
    float*    areas = (float*)(smem + AR_OFF);
    unsigned* keep  = (unsigned*)(smem + KEEP_OFF);
    float*    hist  = (float*)(smem + HIST_OFF);
    float*    redv  = (float*)(smem + REDV_OFF);
    int*      redi  = (int*)(smem + REDI_OFF);
    unsigned* ctrl  = (unsigned*)(smem + CTRL_OFF);

    const int c    = blockIdx.x;
    const int tid  = threadIdx.x;
    const int lane = tid & 31;   // wave32
    const int wave = tid >> 5;   // 8 waves

    if (tid == 0) ctrl[4] = 0u;
    for (int i = tid; i < 313; i += 256) keep[i] = 0u;
    __syncthreads();

    // ---- stage scores + boxes + areas into LDS; column max for col_valid ----
    unsigned localmax = 0u;
    for (int i = tid; i < PAD_N; i += 256) {
        if (i < N_ROIS) {
            const float s = scores[i * N_CLS + c];
            sc[i] = s;
            const unsigned u = __float_as_uint(s);   // softmax scores > 0: bits monotonic
            if (u > localmax) localmax = u;
            const float4 b4 = ((const float4*)boxes)[i * N_CLS + c];
            bxf[i * 4 + 0] = b4.x; bxf[i * 4 + 1] = b4.y;
            bxf[i * 4 + 2] = b4.z; bxf[i * 4 + 3] = b4.w;
            areas[i] = (b4.z - b4.x + 1.0f) * (b4.w - b4.y + 1.0f);
        } else {
            sc[i] = 0.0f;                            // pad: bits == 0
        }
    }
    atomicMax(&ctrl[4], localmax);
    __syncthreads();

    // ---- radix select: bits of the 6000th-largest score, 4 bits/pass ----
    // Histogram-as-matmul: D(16x16) += Ones(16x32) x OneHot(32x16); every D row = histogram.
    // B layout (32x16 f16): lanes 0-15 hold K=0..15, lanes 16-31 hold K=16..31; N = lane&15.
    // Indicator folds prefix+digit into one compare: (u >> shift) == (prefix<<4 | bucket).
    if (tid == 0) { ctrl[0] = 0u; ((float*)ctrl)[1] = (float)PRE_NMS_TOPN; }
    __syncthreads();

    v16h ones;
    #pragma unroll
    for (int j = 0; j < 16; ++j) ones[j] = (_Float16)1.0f;
    const unsigned nsel  = (unsigned)(lane & 15);
    const int      kbase = lane & 16;
    const float4*  scv   = (const float4*)sc;

    for (int p = 7; p >= 0; --p) {
        if (tid < 16) hist[tid] = 0.0f;
        __syncthreads();
        const unsigned prefix = ctrl[0];
        const int      shift  = p * 4;
        const unsigned target = (prefix << 4) | nsel;
        v8f acc0 = {}, acc1 = {};
        for (int base = wave * 64; base < PAD_N; base += 512) {
            v16h b0, b1;
            const int t0 = (base + kbase) >> 2;        // float4 index, 16B aligned
            const int t1 = (base + 32 + kbase) >> 2;
            #pragma unroll
            for (int q = 0; q < 4; ++q) {              // 4 x ds_load_b128 per tile
                const float4 f0 = scv[t0 + q];
                b0[4*q+0] = ind_eq(f0.x, shift, target);
                b0[4*q+1] = ind_eq(f0.y, shift, target);
                b0[4*q+2] = ind_eq(f0.z, shift, target);
                b0[4*q+3] = ind_eq(f0.w, shift, target);
                const float4 f1 = scv[t1 + q];
                b1[4*q+0] = ind_eq(f1.x, shift, target);
                b1[4*q+1] = ind_eq(f1.y, shift, target);
                b1[4*q+2] = ind_eq(f1.z, shift, target);
                b1[4*q+3] = ind_eq(f1.w, shift, target);
            }
            // uniform control flow -> EXEC all ones; two accumulators hide WMMA hazards
            acc0 = __builtin_amdgcn_wmma_f32_16x16x32_f16(
                false, ones, false, b0, (short)0, acc0, false, false);
            acc1 = __builtin_amdgcn_wmma_f32_16x16x32_f16(
                false, ones, false, b1, (short)0, acc1, false, false);
        }
        // C/D layout: VGPR0 on lanes 0-15 is (M=0, N=lane) -> bucket 'lane'
        if (lane < 16) atomicAdd(&hist[lane], acc0[0] + acc1[0]);
        __syncthreads();
        if (tid == 0) {
            float rem = ((float*)ctrl)[1];
            unsigned pick = 0u;
            for (int d = 15; d >= 0; --d) {
                float cnt = hist[d];
                if (d == 0 && p == 7) cnt -= PAD_CNT;  // exact pad correction
                if (cnt >= rem) { pick = (unsigned)d; break; }
                rem -= cnt;
            }
            ctrl[0] = (prefix << 4) | pick;
            ((float*)ctrl)[1] = rem;
        }
        __syncthreads();
    }
    const unsigned kth_bits = ctrl[0];

    // ---- deactivate below pre-NMS threshold; seed the local argmax ----
    float bv = NEG_INF; int bi = -1;
    for (int i = tid; i < N_ROIS; i += 256) {          // disjoint strided ownership
        const float v = sc[i];
        if (__float_as_uint(v) < kth_bits) {
            sc[i] = NEG_INF;
        } else if (v > bv) {
            bv = v; bi = i;
        }
    }
    __syncthreads();

    // ---- greedy NMS: fused suppress + next-argmax, 2 barriers/iteration ----
    for (int it = 0; it < POST_NMS_TOPN; ++it) {
        float rv = bv; int ri = bi;
        #pragma unroll
        for (int off = 16; off > 0; off >>= 1) {       // wave32 butterfly
            const float ov = __shfl_down(rv, off);
            const int   oi = __shfl_down(ri, off);
            if (ov > rv) { rv = ov; ri = oi; }
        }
        if (lane == 0) { redv[wave] = rv; redi[wave] = ri; }
        __syncthreads();
        if (tid == 0) {
            float bb = NEG_INF; int bbi = -1;
            for (int w = 0; w < 8; ++w)
                if (redv[w] > bb) { bb = redv[w]; bbi = redi[w]; }
            ((float*)ctrl)[2] = bb;
            ((int*)ctrl)[3]   = bbi;
            if (bbi >= 0 && bb > NEG_INF) keep[bbi >> 5] |= (1u << (bbi & 31));
        }
        __syncthreads();
        const float bestv = ((float*)ctrl)[2];
        if (!(bestv > NEG_INF)) break;                 // uniform LDS broadcast
        const int    best  = ((int*)ctrl)[3];
        const float4 B     = bx4[best];
        const float  barea = areas[best];
        bv = NEG_INF; bi = -1;
        for (int i = tid; i < N_ROIS; i += 256) {      // one scan: suppress + re-argmax
            const float v = sc[i];
            if (v > NEG_INF) {
                const float4 bb = bx4[i];
                const float iw = fmaxf(fminf(B.z, bb.z) - fmaxf(B.x, bb.x) + 1.0f, 0.0f);
                const float ih = fmaxf(fminf(B.w, bb.w) - fmaxf(B.y, bb.y) + 1.0f, 0.0f);
                const float inter = iw * ih;
                const float iou   = inter / (barea + areas[i] - inter);
                if (iou > NMS_THRESH) sc[i] = NEG_INF; // self box suppressed (iou == 1)
                else if (v > bv) { bv = v; bi = i; }
            }
        }
        // no barrier needed for sc: each thread reads/writes only its own stride set
    }

    // ---- write this class's column of dists_all ----
    const bool col_valid = (c != 0) && (__uint_as_float(ctrl[4]) > SCORE_THRESH);
    for (int i = tid; i < N_ROIS; i += 256) {
        const bool kb = (keep[i >> 5] >> (i & 31)) & 1u;
        const float s = scores[i * N_CLS + c];
        dists_out[i * N_CLS + c] = (kb && col_valid) ? s : 0.0f;
    }
}

// =====================================================================
// Kernel 2: per-ROI max / argmax over classes (dists_all >= 0, so init 0/label 0)
// =====================================================================
__global__ __launch_bounds__(256) void rowmax_kernel(const float* __restrict__ dists,
                                                     float* __restrict__ spre,
                                                     int* __restrict__ lpre)
{
    const int i = blockIdx.x * blockDim.x + threadIdx.x;
    if (i >= N_ROIS) return;
    const float* row = dists + (size_t)i * N_CLS;
    float m = 0.0f; int lab = 0;
    for (int c = 0; c < N_CLS; ++c) {
        const float v = row[c];
        if (v > m) { m = v; lab = c; }
    }
    spre[i] = m;
    lpre[i] = lab;
}

// =====================================================================
// Kernel 3: single-block iterative top-64 (threshold + validity), outputs as f32
// =====================================================================
__global__ __launch_bounds__(1024) void topk_kernel(const float* __restrict__ spre,
                                                    const int* __restrict__ lpre,
                                                    float* __restrict__ out)
{
    __shared__ float sc[N_ROIS];
    __shared__ float redv[32];
    __shared__ int   redi[32];
    const int tid  = threadIdx.x;
    const int lane = tid & 31;
    const int wave = tid >> 5;

    for (int i = tid; i < N_ROIS; i += 1024) {
        const float v = spre[i];
        sc[i] = (v > SCORE_THRESH) ? v : NEG_INF;
    }
    __syncthreads();

    const size_t OFF = (size_t)N_ROIS * N_CLS;
    for (int j = 0; j < MAX_PER_IMG; ++j) {
        float bv = NEG_INF; int bi = -1;
        for (int i = tid; i < N_ROIS; i += 1024) {
            const float v = sc[i];
            if (v > bv) { bv = v; bi = i; }
        }
        #pragma unroll
        for (int off = 16; off > 0; off >>= 1) {
            const float ov = __shfl_down(bv, off);
            const int   oi = __shfl_down(bi, off);
            if (ov > bv) { bv = ov; bi = oi; }
        }
        if (lane == 0) { redv[wave] = bv; redi[wave] = bi; }
        __syncthreads();
        if (tid == 0) {
            float bb = NEG_INF; int bbi = -1;
            for (int w = 0; w < 32; ++w)
                if (redv[w] > bb) { bb = redv[w]; bbi = redi[w]; }
            const bool valid = (bbi >= 0) && (bb > NEG_INF);
            out[OFF + j]                    = valid ? bb : 0.0f;
            out[OFF + MAX_PER_IMG + j]      = valid ? (float)bbi : -1.0f;
            out[OFF + 2 * MAX_PER_IMG + j]  = valid ? (float)lpre[bbi] : -1.0f;
            if (valid) sc[bbi] = NEG_INF;
        }
        __syncthreads();
    }
}

// =====================================================================
extern "C" void kernel_launch(void* const* d_in, const int* in_sizes, int n_in,
                              void* d_out, int out_size, void* d_ws, size_t ws_size,
                              hipStream_t stream)
{
    const float* scores = (const float*)d_in[0];   // [10000, 151]
    const float* boxes  = (const float*)d_in[1];   // [10000, 151, 4]
    float* out = (float*)d_out;                    // dists_all | top_scores | top_inds | top_labels
    float* spre = (float*)d_ws;                                  // 10000 f32
    int*   lpre = (int*)((char*)d_ws + N_ROIS * sizeof(float));  // 10000 i32

    nms_per_class<<<N_CLS, 256, LDS_BYTES, stream>>>(scores, boxes, out);
    rowmax_kernel<<<(N_ROIS + 255) / 256, 256, 0, stream>>>(out, spre, lpre);
    topk_kernel<<<1, 1024, 0, stream>>>(spre, lpre, out);
}
